// BoundaryCorrectionModule_86887188398375
// MI455X (gfx1250) — matrix operand order repaired and checked by reading, not compiled, not run on hardware
//
#include <hip/hip_runtime.h>
#include <hip/hip_bf16.h>
#include <stddef.h>

#define B_ROWS 16384
#define D_DIM  1024
#define NELEM  (B_ROWS * D_DIM)

typedef __attribute__((ext_vector_type(16))) __bf16 bf16x16;
typedef __attribute__((ext_vector_type(8)))  __bf16 bf16x8;
typedef __attribute__((ext_vector_type(8)))  float  f32x8;

static __device__ __forceinline__ unsigned short f2bf(float f) {
  unsigned u = __float_as_uint(f);
  u += 0x7FFFu + ((u >> 16) & 1u);   // round-to-nearest-even
  return (unsigned short)(u >> 16);
}

static __device__ __forceinline__ float sigm(float x) {
  return 1.0f / (1.0f + __expf(-x));
}

// ---------------------------------------------------------------------------
// f32 -> bf16 weight conversion
// ---------------------------------------------------------------------------
__global__ void k_cvt_bf16(const float* __restrict__ in,
                           unsigned short* __restrict__ out, int n) {
  int i = blockIdx.x * 256 + threadIdx.x;
  if (i < n) out[i] = f2bf(in[i]);
}

// ---------------------------------------------------------------------------
// Multi-term GEMM:  C[B_ROWS, D_DIM] = sum_t  A_t[B_ROWS, D_DIM] @ W_t^T
// A_t: bf16 row-major, K contiguous (K = D_DIM per term)
// W_t: bf16, row n is K-contiguous with compile-time row stride LDW
//      (pointer is pre-offset to the term's K-slice)
// One workgroup = 8 waves (2 along M x 4 along N) = 128x256 C tile.
// One wave = 64x64 C tile = 4x4 WMMA tiles: 16 wmma per K-step fed by
// 8 A-loads + 8 B-loads, all immediate-offset b128s (1.0 loads per WMMA).
// ---------------------------------------------------------------------------
template <int NT, int LDW>
__global__ __launch_bounds__(256)
void k_gemm(const unsigned short* __restrict__ A0, const unsigned short* __restrict__ W0,
            const unsigned short* __restrict__ A1, const unsigned short* __restrict__ W1,
            const unsigned short* __restrict__ A2, const unsigned short* __restrict__ W2,
            float* __restrict__ C) {
  const int lane = threadIdx.x & 31;
  const int wave = threadIdx.x >> 5;
  const int lmod = lane & 15;
  const int half = lane >> 4;
  const int m0 = blockIdx.x * 128 + (wave & 1) * 64;   // 4 M-tiles of 16
  const int n0 = blockIdx.y * 256 + (wave >> 1) * 64;  // 4 N-tiles of 16

  const f32x8 vzero = {0.f, 0.f, 0.f, 0.f, 0.f, 0.f, 0.f, 0.f};
  f32x8 acc[4][4];
#pragma unroll
  for (int i = 0; i < 4; ++i)
#pragma unroll
    for (int j = 0; j < 4; ++j) acc[i][j] = vzero;

  const unsigned short* As[3] = {A0, A1, A2};
  const unsigned short* Ws[3] = {W0, W1, W2};

#pragma unroll
  for (int t = 0; t < NT; ++t) {
    // A fragment lanes: row = m0 + lmod (+16*i), K chunks at half*8 / half*8+16
    const unsigned short* pa = As[t] + (size_t)(m0 + lmod) * D_DIM + half * 8;
    // B fragment lanes: col = n0 + lmod (+16*j), K chunks at half*8 / half*8+16
    const unsigned short* pb = Ws[t] + (size_t)(n0 + lmod) * LDW + half * 8;

#pragma unroll 1
    for (int k0 = 0; k0 < D_DIM; k0 += 32) {
      bf16x16 fa[4];
      bf16x16 fb[4];
#pragma unroll
      for (int i = 0; i < 4; ++i) {
        const unsigned short* p = pa + i * 16 * D_DIM + k0;   // imm offsets
        bf16x8 lo = *(const bf16x8*)(const void*)p;
        bf16x8 hi = *(const bf16x8*)(const void*)(p + 16);
        fa[i] = __builtin_shufflevector(lo, hi, 0, 1, 2, 3, 4, 5, 6, 7,
                                        8, 9, 10, 11, 12, 13, 14, 15);
      }
#pragma unroll
      for (int j = 0; j < 4; ++j) {
        const unsigned short* q = pb + j * 16 * LDW + k0;  // imm offsets
        bf16x8 blo = *(const bf16x8*)(const void*)q;
        bf16x8 bhi = *(const bf16x8*)(const void*)(q + 16);
        fb[j] = __builtin_shufflevector(blo, bhi, 0, 1, 2, 3, 4, 5, 6, 7,
                                        8, 9, 10, 11, 12, 13, 14, 15);
      }
#pragma unroll
      for (int i = 0; i < 4; ++i) {
#pragma unroll
        for (int j = 0; j < 4; ++j) {
          acc[i][j] = __builtin_amdgcn_wmma_f32_16x16x32_bf16(
              false, fa[i], false, fb[j], (short)0, acc[i][j], false, false);
        }
      }
    }
  }

  // C/D layout: VGPR r -> row (half*8 + r) of the 16x16 tile, col = lmod
#pragma unroll
  for (int i = 0; i < 4; ++i) {
#pragma unroll
    for (int j = 0; j < 4; ++j) {
      const int col = n0 + j * 16 + lmod;
#pragma unroll
      for (int r = 0; r < 8; ++r) {
        const int row = m0 + i * 16 + half * 8 + r;
        C[(size_t)row * D_DIM + col] = acc[i][j][r];
      }
    }
  }
}

// ---------------------------------------------------------------------------
// Elementwise epilogues (bias + activations + bf16 re-quantization fused)
// Grids are exact (NELEM/256), no bounds checks needed.
// ---------------------------------------------------------------------------
__global__ void k_init(const float* __restrict__ hp, const float* __restrict__ hn,
                       float* __restrict__ Sf, float* __restrict__ Mf, float* __restrict__ Df,
                       unsigned short* __restrict__ Sb, unsigned short* __restrict__ Mb,
                       unsigned short* __restrict__ Db) {
  int i = blockIdx.x * 256 + threadIdx.x;
  float a = hp[i], b = hn[i];
  float s = 0.5f * (a + b);
  Sf[i] = s; Mf[i] = a; Df[i] = b;
  Sb[i] = f2bf(s); Mb[i] = f2bf(a); Db[i] = f2bf(b);
}

// out = tanh(pre_t) * sigmoid(pre_g + bias)   (S_new / D_new)
__global__ void k_gate(const float* __restrict__ pre_t, const float* __restrict__ pre_g,
                       const float* __restrict__ bias,
                       float* __restrict__ outF, unsigned short* __restrict__ outB) {
  int i = blockIdx.x * 256 + threadIdx.x;
  float x = tanhf(pre_t[i]) * sigm(pre_g[i] + bias[i & (D_DIM - 1)]);
  outF[i] = x; outB[i] = f2bf(x);
}

// RB = bf16( sigmoid(pre_r + b_Mr) * M )
__global__ void k_rmul(const float* __restrict__ pre_r, const float* __restrict__ bias,
                       const float* __restrict__ Mf, unsigned short* __restrict__ RB) {
  int i = blockIdx.x * 256 + threadIdx.x;
  float r = sigm(pre_r[i] + bias[i & (D_DIM - 1)]);
  RB[i] = f2bf(r * Mf[i]);
}

// M_new = (1-z)*M + z*tanh(pre_h + b_Mh),  z = sigmoid(pre_z + b_Mz)
__global__ void k_mnew(const float* __restrict__ pre_z, const float* __restrict__ bz,
                       const float* __restrict__ pre_h, const float* __restrict__ bh,
                       float* __restrict__ Mf, unsigned short* __restrict__ Mb) {
  int i = blockIdx.x * 256 + threadIdx.x;
  int c = i & (D_DIM - 1);
  float z = sigm(pre_z[i] + bz[c]);
  float h = tanhf(pre_h[i] + bh[c]);
  float m = (1.0f - z) * Mf[i] + z * h;
  Mf[i] = m; Mb[i] = f2bf(m);
}

__global__ void k_final(const float* __restrict__ Mf, const float* __restrict__ Sf,
                        const float* __restrict__ Df, const float* __restrict__ rs,
                        float* __restrict__ out) {
  int i = blockIdx.x * 256 + threadIdx.x;
  out[i] = Mf[i] + rs[0] * (Sf[i] + Df[i]);
}

// ---------------------------------------------------------------------------
extern "C" void kernel_launch(void* const* d_in, const int* in_sizes, int n_in,
                              void* d_out, int out_size, void* d_ws, size_t ws_size,
                              hipStream_t stream) {
  (void)in_sizes; (void)n_in; (void)out_size; (void)ws_size;
  const float* h_prev = (const float*)d_in[0];
  const float* h_next = (const float*)d_in[1];
  const float* W_SS = (const float*)d_in[2];
  const float* W_SM = (const float*)d_in[3];
  const float* W_SD = (const float*)d_in[4];
  const float* W_Mz = (const float*)d_in[5];
  const float* b_Mz = (const float*)d_in[6];
  const float* W_Mr = (const float*)d_in[7];
  const float* b_Mr = (const float*)d_in[8];
  const float* W_Mh = (const float*)d_in[9];
  const float* b_Mh = (const float*)d_in[10];
  const float* W_DS = (const float*)d_in[11];
  const float* W_DM = (const float*)d_in[12];
  const float* W_DD = (const float*)d_in[13];
  const float* W_gS = (const float*)d_in[14];
  const float* b_gS = (const float*)d_in[15];
  const float* W_gD = (const float*)d_in[16];
  const float* b_gD = (const float*)d_in[17];
  const float* rscale = (const float*)d_in[18];

  char* ws = (char*)d_ws;
  size_t off = 0;
  auto take = [&](size_t bytes) -> void* {
    void* r = (void*)(ws + off);
    off += (bytes + 255) & ~(size_t)255;
    return r;
  };
  const size_t D2 = (size_t)D_DIM * D_DIM;

  // bf16 weight copies (re-converted every call: deterministic, ~114MB traffic)
  unsigned short* wSS = (unsigned short*)take(D2 * 2);
  unsigned short* wSM = (unsigned short*)take(D2 * 2);
  unsigned short* wSD = (unsigned short*)take(D2 * 2);
  unsigned short* wMz = (unsigned short*)take(3 * D2 * 2);
  unsigned short* wMr = (unsigned short*)take(3 * D2 * 2);
  unsigned short* wMh = (unsigned short*)take(3 * D2 * 2);
  unsigned short* wDS = (unsigned short*)take(D2 * 2);
  unsigned short* wDM = (unsigned short*)take(D2 * 2);
  unsigned short* wDD = (unsigned short*)take(D2 * 2);
  unsigned short* wgS = (unsigned short*)take(2 * D2 * 2);
  unsigned short* wgD = (unsigned short*)take(2 * D2 * 2);

  // f32 state + scratch
  float* Sf = (float*)take((size_t)NELEM * 4);
  float* Mf = (float*)take((size_t)NELEM * 4);
  float* Df = (float*)take((size_t)NELEM * 4);
  float* T0 = (float*)take((size_t)NELEM * 4);
  float* T1 = (float*)d_out;  // reuse output buffer as second f32 scratch

  // bf16 activation mirrors (GEMM A operands)
  unsigned short* Sb = (unsigned short*)take((size_t)NELEM * 2);
  unsigned short* Mb = (unsigned short*)take((size_t)NELEM * 2);
  unsigned short* Db = (unsigned short*)take((size_t)NELEM * 2);
  unsigned short* RB = (unsigned short*)take((size_t)NELEM * 2);

  auto cvt = [&](const float* src, unsigned short* dst, size_t n) {
    k_cvt_bf16<<<dim3((unsigned)((n + 255) / 256)), dim3(256), 0, stream>>>(src, dst, (int)n);
  };
  cvt(W_SS, wSS, D2); cvt(W_SM, wSM, D2); cvt(W_SD, wSD, D2);
  cvt(W_Mz, wMz, 3 * D2); cvt(W_Mr, wMr, 3 * D2); cvt(W_Mh, wMh, 3 * D2);
  cvt(W_DS, wDS, D2); cvt(W_DM, wDM, D2); cvt(W_DD, wDD, D2);
  cvt(W_gS, wgS, 2 * D2); cvt(W_gD, wgD, 2 * D2);

  const dim3 gG(B_ROWS / 128, D_DIM / 256), bG(256);
  const dim3 gE(NELEM / 256), bE(256);

  k_init<<<gE, bE, 0, stream>>>(h_prev, h_next, Sf, Mf, Df, Sb, Mb, Db);

  for (int step = 0; step < 2; ++step) {
    // S_input = S@W_SS^T + M@W_SM^T + D@W_SD^T              (LDW = D)
    k_gemm<3, D_DIM><<<gG, bG, 0, stream>>>(Sb, wSS, Mb, wSM, Db, wSD, T0);
    // S_gate pre = [M,D] @ W_gS^T  (K-concat -> 2 terms, LDW = 2D)
    k_gemm<2, 2 * D_DIM><<<gG, bG, 0, stream>>>(Mb, wgS, Db, wgS + D_DIM,
                                                nullptr, nullptr, T1);
    k_gate<<<gE, bE, 0, stream>>>(T0, T1, b_gS, Sf, Sb);  // S_new

    // z pre, r pre = [S_new,M,D] @ W_Mz/Mr^T               (LDW = 3D)
    k_gemm<3, 3 * D_DIM><<<gG, bG, 0, stream>>>(Sb, wMz, Mb, wMz + D_DIM,
                                                Db, wMz + 2 * D_DIM, T0);
    k_gemm<3, 3 * D_DIM><<<gG, bG, 0, stream>>>(Sb, wMr, Mb, wMr + D_DIM,
                                                Db, wMr + 2 * D_DIM, T1);
    k_rmul<<<gE, bE, 0, stream>>>(T1, b_Mr, Mf, RB);      // RB = bf16(r*M)

    // h_cand pre = [S_new, r*M, D] @ W_Mh^T                (LDW = 3D)
    k_gemm<3, 3 * D_DIM><<<gG, bG, 0, stream>>>(Sb, wMh, RB, wMh + D_DIM,
                                                Db, wMh + 2 * D_DIM, T1);
    k_mnew<<<gE, bE, 0, stream>>>(T0, b_Mz, T1, b_Mh, Mf, Mb);  // M_new

    // D_input = S_new@W_DS^T + M_new@W_DM^T + D@W_DD^T     (LDW = D)
    k_gemm<3, D_DIM><<<gG, bG, 0, stream>>>(Sb, wDS, Mb, wDM, Db, wDD, T0);
    // D_gate pre = [S_new, M_new] @ W_gD^T                 (LDW = 2D)
    k_gemm<2, 2 * D_DIM><<<gG, bG, 0, stream>>>(Sb, wgD, Mb, wgD + D_DIM,
                                                nullptr, nullptr, T1);
    k_gate<<<gE, bE, 0, stream>>>(T0, T1, b_gD, Df, Db);  // D_new
  }

  k_final<<<gE, bE, 0, stream>>>(Mf, Sf, Df, rscale, (float*)d_out);
}
